// MarginNet_74809740361874
// MI455X (gfx1250) — matrix operand order repaired and checked by reading, not compiled, not run on hardware
//
#include <hip/hip_runtime.h>
#include <hip/hip_bf16.h>

typedef __attribute__((ext_vector_type(16))) _Float16 v16h;
typedef __attribute__((ext_vector_type(8)))  _Float16 v8h;
typedef __attribute__((ext_vector_type(8)))  float    v8f;
typedef __attribute__((ext_vector_type(4)))  float    v4f;

// Problem constants (from reference: n=8192, feat=512, d=128, k=8)
#define N_ROWS   8192
#define FEAT     512
#define EMB      128
#define BATCH_K  8
#define NSAMP    7                      // k-1
#define NANCH    (N_ROWS * NSAMP)       // 57344
#define OFF1     (NANCH)                            // xe[a_idx]
#define OFF2     (OFF1 + NANCH * EMB)               // xe[p_idx]
#define OFF3     (OFF2 + NANCH * EMB)               // xe[n_idx]
#define OFF4     (OFF3 + NANCH * EMB)               // xe

#define CUTOFF_D   0.5f
#define NZ_CUT     1.4f

static __device__ __forceinline__ v8f wmma16(v16h a, v16h b, v8f c) {
  return __builtin_amdgcn_wmma_f32_16x16x32_f16(
      /*neg_a=*/false, a, /*neg_b=*/false, b,
      /*c_mod=*/(short)0, c, /*reuse_a=*/false, /*reuse_b=*/false);
}

// monotone float<->uint encode for atomic min/max (order-independent => deterministic)
static __device__ __forceinline__ unsigned enc_f(float f) {
  unsigned b = __float_as_uint(f);
  return (b & 0x80000000u) ? ~b : (b | 0x80000000u);
}
static __device__ __forceinline__ float dec_f(unsigned ue) {
  unsigned b = (ue & 0x80000000u) ? (ue & 0x7FFFFFFFu) : ~ue;
  return __uint_as_float(b);
}

static __device__ __forceinline__ float gumbel_hash(unsigned i, unsigned j, unsigned s) {
  unsigned h = i * 0x9E3779B1u ^ j * 0x85EBCA77u ^ (s + 1u) * 0xC2B2AE3Du;
  h ^= h >> 16; h *= 0x7FEB352Du; h ^= h >> 15; h *= 0x846CA68Bu; h ^= h >> 16;
  float u = (float)(h >> 8) * (1.0f / 16777216.0f) + 1e-7f;
  return -__logf(-__logf(u));
}

// A-fragment loader for 16-bit 16x32 tile, f16 source, contiguous row.
// rowp points at (row base + kstep + hi*16); contiguous K layout k = 16*hi + e.
static __device__ __forceinline__ v16h load_afrag_h(const _Float16* rowp) {
  v8h lo = *(const v8h*)(rowp);
  v8h hh = *(const v8h*)(rowp + 8);
  v16h a;
#pragma unroll
  for (int e = 0; e < 8; ++e) { a[e] = lo[e]; a[8 + e] = hh[e]; }
  return a;
}

// ---------------- Prep: Wt[n][k] = (f16) W[k][n]   (128 x 512, 128 KB, L2-resident)
__global__ __launch_bounds__(256) void k_prep_w(
    const float* __restrict__ W, _Float16* __restrict__ Wt)
{
  int g = blockIdx.x * 256 + threadIdx.x;          // 0 .. 128*512-1
  if (g >= EMB * FEAT) return;
  int n = g >> 9;              // /512
  int k = g & (FEAT - 1);
  Wt[n * FEAT + k] = (_Float16)W[k * EMB + n];
}

// ---------------- Kernel 1: emb = x@W + b, L2-normalize, emit xe(f32) + xe_h(f16)
__global__ __launch_bounds__(256) void k_emb_norm(
    const float* __restrict__ x, const _Float16* __restrict__ Wt,
    const float* __restrict__ bias, float* __restrict__ xe,
    _Float16* __restrict__ xeh)
{
  __shared__ float tile[16 * EMB];
  __shared__ float rinv[16];
  const int tid  = threadIdx.x;
  const int wave = tid >> 5;
  const int lane = tid & 31;
  const int mL   = lane & 15;
  const int hi   = lane >> 4;
  const int i0   = blockIdx.x * 16;
  const int n0   = wave * 16;

  v8f c = {};
  for (int ks = 0; ks < FEAT; ks += 32) {
    // A fragment: row i0+mL of x (f32 -> f16). Contiguous 16 per lane.
    const float* xr = x + (size_t)(i0 + mL) * FEAT + ks + hi * 16;
    v4f p0 = *(const v4f*)(xr);
    v4f p1 = *(const v4f*)(xr + 4);
    v4f p2 = *(const v4f*)(xr + 8);
    v4f p3 = *(const v4f*)(xr + 12);
    v16h a;
#pragma unroll
    for (int e = 0; e < 4; ++e) {
      a[e]      = (_Float16)p0[e];
      a[4 + e]  = (_Float16)p1[e];
      a[8 + e]  = (_Float16)p2[e];
      a[12 + e] = (_Float16)p3[e];
    }
    // B fragment: column n0+mL of W == contiguous row of Wt -> one 32B load
    v16h b = *(const v16h*)(Wt + (size_t)(n0 + mL) * FEAT + ks + hi * 16);
    c = wmma16(a, b, c);
  }

  const float bv = bias[n0 + mL];
#pragma unroll
  for (int v = 0; v < 8; ++v) {
    int m = v + hi * 8;
    tile[m * EMB + n0 + mL] = c[v] + bv;
  }
  __syncthreads();
  if (tid < 16) {
    float s = 0.0f;
    for (int cc = 0; cc < EMB; ++cc) { float t = tile[tid * EMB + cc]; s += t * t; }
    rinv[tid] = 1.0f / fmaxf(sqrtf(s), 1e-12f);
  }
  __syncthreads();
  for (int e = tid; e < 16 * EMB; e += 256) {
    int m = e >> 7, cc = e & (EMB - 1);
    float val = tile[e] * rinv[m];
    xe[(size_t)(i0 + m) * EMB + cc]  = val;
    xeh[(size_t)(i0 + m) * EMB + cc] = (_Float16)val;
  }
}

// ---------------- init encoded global min/max
__global__ void k_init(unsigned* gmm) {
  gmm[0] = 0xFF800000u;  // enc(+inf) -> running min
  gmm[1] = 0x007FFFFFu;  // enc(-inf) -> running max
}

// ---------------- Kernel 2: sim row-band GEMM + global min/max of log_w
__global__ __launch_bounds__(256) void k_minmax(
    const _Float16* __restrict__ xeh, unsigned* __restrict__ gmm)
{
  const int tid  = threadIdx.x;
  const int wave = tid >> 5;
  const int lane = tid & 31;
  const int mL   = lane & 15;
  const int hi   = lane >> 4;
  const int i0   = blockIdx.x * 16;

  // preload A fragments for this 16-row band (reused for all 512 column tiles)
  v16h af[4];
#pragma unroll
  for (int kk = 0; kk < 4; ++kk)
    af[kk] = load_afrag_h(xeh + (size_t)(i0 + mL) * EMB + kk * 32 + hi * 16);

  float lmin = INFINITY, lmax = -INFINITY;
  for (int jt = wave; jt < N_ROWS / 16; jt += 8) {
    const int j0 = jt * 16;
    v8f c = {};
#pragma unroll
    for (int kk = 0; kk < 4; ++kk) {
      v16h b = *(const v16h*)(xeh + (size_t)(j0 + mL) * EMB + kk * 32 + hi * 16);
      c = wmma16(af[kk], b, c);
    }
#pragma unroll
    for (int v = 0; v < 8; ++v) {
      int i = i0 + v + hi * 8;
      int j = j0 + mL;
      float d2 = 2.0f - 2.0f * c[v] + ((i == j) ? 1.0f : 0.0f);
      float dist = fmaxf(sqrtf(fmaxf(d2, 0.0f)), CUTOFF_D);
      float lw = -126.0f * __logf(dist)
                 - 62.5f * __logf(fmaxf(1.0f - 0.25f * dist * dist, 1e-8f));
      lmin = fminf(lmin, lw);
      lmax = fmaxf(lmax, lw);
    }
  }
  __shared__ float smin[256], smax[256];
  smin[tid] = lmin; smax[tid] = lmax;
  __syncthreads();
  for (int s = 128; s > 0; s >>= 1) {
    if (tid < s) {
      smin[tid] = fminf(smin[tid], smin[tid + s]);
      smax[tid] = fmaxf(smax[tid], smax[tid + s]);
    }
    __syncthreads();
  }
  if (tid == 0) {
    atomicMin(&gmm[0], enc_f(smin[0]));
    atomicMax(&gmm[1], enc_f(smax[0]));
  }
}

// ---------------- Kernel 3: recompute sim, Gumbel-max categorical sampling
__global__ __launch_bounds__(256) void k_sample(
    const _Float16* __restrict__ xeh, const unsigned* __restrict__ gmm,
    int* __restrict__ nidx)
{
  __shared__ float tile[16 * 128];
  __shared__ float rbv[16 * 16 * NSAMP];
  __shared__ int   rbi[16 * 16 * NSAMP];

  const int tid  = threadIdx.x;
  const int wave = tid >> 5;
  const int lane = tid & 31;
  const int mL   = lane & 15;
  const int hi   = lane >> 4;
  const int i0   = blockIdx.x * 16;

  const float gmin = dec_f(gmm[0]);
  const float gmax = dec_f(gmm[1]);
  const float inv  = 1.0f / (gmax - gmin + 1e-8f);

  v16h af[4];
#pragma unroll
  for (int kk = 0; kk < 4; ++kk)
    af[kk] = load_afrag_h(xeh + (size_t)(i0 + mL) * EMB + kk * 32 + hi * 16);

  // per-thread row-partial argmax state (row = tid>>4, column residue = tid&15)
  const int prow = tid >> 4;
  const int pcl  = tid & 15;
  const int gi   = i0 + prow;
  const int myblk = gi >> 3;
  float bv[NSAMP]; int bi[NSAMP];
#pragma unroll
  for (int s = 0; s < NSAMP; ++s) { bv[s] = -INFINITY; bi[s] = 0; }

  for (int step = 0; step < N_ROWS / 128; ++step) {
    const int jbase = step * 128;
    const int j0 = jbase + wave * 16;
    v8f c = {};
#pragma unroll
    for (int kk = 0; kk < 4; ++kk) {
      v16h b = *(const v16h*)(xeh + (size_t)(j0 + mL) * EMB + kk * 32 + hi * 16);
      c = wmma16(af[kk], b, c);
    }
    __syncthreads();  // previous iteration's readers are done
#pragma unroll
    for (int v = 0; v < 8; ++v)
      tile[(v + hi * 8) * 128 + wave * 16 + mL] = c[v];
    __syncthreads();

#pragma unroll
    for (int q = 0; q < 8; ++q) {
      const int jl = pcl + q * 16;
      const int gj = jbase + jl;
      if ((gj >> 3) == myblk) continue;               // same-class block: masked out
      float sim = tile[prow * 128 + jl];
      float d2 = 2.0f - 2.0f * sim + ((gi == gj) ? 1.0f : 0.0f);
      float dist = fmaxf(sqrtf(fmaxf(d2, 0.0f)), CUTOFF_D);
      float lw = -126.0f * __logf(dist)
                 - 62.5f * __logf(fmaxf(1.0f - 0.25f * dist * dist, 1e-8f));
      float lwn = (lw - gmin) * inv;                  // in [0,1]
      // valid (weight>0) entries dominate; bare-mask entries form uniform fallback
      float bonus = (dist < NZ_CUT) ? (lwn + 1000.0f) : 0.0f;
#pragma unroll
      for (int s = 0; s < NSAMP; ++s) {
        float score = gumbel_hash((unsigned)gi, (unsigned)gj, (unsigned)s) + bonus;
        if (score > bv[s]) { bv[s] = score; bi[s] = gj; }
      }
    }
  }

  __syncthreads();
#pragma unroll
  for (int s = 0; s < NSAMP; ++s) {
    rbv[(prow * 16 + pcl) * NSAMP + s] = bv[s];
    rbi[(prow * 16 + pcl) * NSAMP + s] = bi[s];
  }
  __syncthreads();
  if (pcl == 0) {
    for (int s = 0; s < NSAMP; ++s) {
      float best = -INFINITY; int besti = 0;
      for (int t = 0; t < 16; ++t) {
        float vv = rbv[(prow * 16 + t) * NSAMP + s];
        if (vv > best) { best = vv; besti = rbi[(prow * 16 + t) * NSAMP + s]; }
      }
      nidx[gi * NSAMP + s] = besti;
    }
  }
}

// ---------------- Kernel 4: gather tuple outputs
__global__ __launch_bounds__(256) void k_gather(
    const float* __restrict__ xe, const int* __restrict__ nidx,
    float* __restrict__ out)
{
  int g = blockIdx.x * 256 + threadIdx.x;
  if (g >= NANCH * EMB) return;
  int t = g >> 7;
  int c = g & (EMB - 1);
  int i = t / NSAMP;
  int s = t - i * NSAMP;
  int base = i & ~(BATCH_K - 1);
  int r = i & (BATCH_K - 1);
  int pi = base + s + ((s >= r) ? 1 : 0);
  int ni = nidx[t];
  out[OFF1 + (size_t)t * EMB + c] = xe[(size_t)i * EMB + c];
  out[OFF2 + (size_t)t * EMB + c] = xe[(size_t)pi * EMB + c];
  out[OFF3 + (size_t)t * EMB + c] = xe[(size_t)ni * EMB + c];
  if (c == 0) out[t] = (float)i;
}

extern "C" void kernel_launch(void* const* d_in, const int* in_sizes, int n_in,
                              void* d_out, int out_size, void* d_ws, size_t ws_size,
                              hipStream_t stream) {
  const float* x  = (const float*)d_in[0];
  const float* W  = (const float*)d_in[1];
  const float* b  = (const float*)d_in[2];
  (void)in_sizes; (void)n_in; (void)out_size; (void)ws_size;

  float* out = (float*)d_out;
  float* xe  = out + OFF4;                    // xe lives in its final output slot

  // workspace layout (64B-aligned sections)
  char* ws = (char*)d_ws;
  _Float16* xeh  = (_Float16*)ws;                                        // 2 MB
  unsigned* gmm  = (unsigned*)(ws + (size_t)N_ROWS * EMB * 2);           // 8 B
  int*      nidx = (int*)(ws + (size_t)N_ROWS * EMB * 2 + 64);           // 224 KB
  _Float16* Wt   = (_Float16*)(ws + (size_t)N_ROWS * EMB * 2 + 64 +
                               (size_t)NANCH * 4);                       // 128 KB

  k_prep_w<<<(EMB * FEAT + 255) / 256, 256, 0, stream>>>(W, Wt);
  k_init<<<1, 1, 0, stream>>>(gmm);
  k_emb_norm<<<N_ROWS / 16, 256, 0, stream>>>(x, Wt, b, xe, xeh);
  k_minmax<<<N_ROWS / 16, 256, 0, stream>>>(xeh, gmm);
  k_sample<<<N_ROWS / 16, 256, 0, stream>>>(xeh, gmm, nidx);
  k_gather<<<(NANCH * EMB + 255) / 256, 256, 0, stream>>>(xe, nidx, out);
}
